// CrossAttentionBlock_26499948216616
// MI455X (gfx1250) — compile-verified, matrix operand order checked by reading, not run on hardware
//
#include <hip/hip_runtime.h>

#define BATCH 2
#define SQ 2048
#define SKV 2048
#define DDIM 1024
#define NH 16
#define HD 64

typedef __attribute__((ext_vector_type(16))) __bf16 v16bf;
typedef __attribute__((ext_vector_type(8)))  float  v8f;

__device__ __forceinline__ unsigned short f32_to_bf16(float f) {
    unsigned u = __builtin_bit_cast(unsigned, f);
    u += 0x7FFFu + ((u >> 16) & 1u);          // round-to-nearest-even
    return (unsigned short)(u >> 16);
}
__device__ __forceinline__ float bf16_to_f32(unsigned short h) {
    unsigned u = ((unsigned)h) << 16;
    return __builtin_bit_cast(float, u);
}

// ---------------------------------------------------------------- convert f32 -> bf16
__global__ void cvt_f32_bf16(const float* __restrict__ in, unsigned short* __restrict__ out, int n) {
    int i = blockIdx.x * 256 + threadIdx.x;
    if (i < n) out[i] = f32_to_bf16(in[i]);
}

// ---------------------------------------------------------------- bf16 GEMM: out[m,e] = sum_d A[m,d]*W[e,d] + bias[e]
// Wave tile 64x64, k-step 32, 16 WMMAs/iter. Block = 128 threads (4 waves).
// One base pointer per matrix; fragment offsets folded into the 24-bit vmem immediate.
__global__ void __launch_bounds__(128, 1)
gemm_bf16(const unsigned short* __restrict__ A,   // [M, 1024] bf16
          const unsigned short* __restrict__ W,   // [1024, 1024] bf16 (row e, contiguous d)
          const float* __restrict__ bias,         // [1024]
          unsigned short* __restrict__ out_bf,    // [M, 1024] or null
          float* __restrict__ out_f32,            // [M, 1024] or null
          int M) {
    const int wave = threadIdx.x >> 5;
    const int lane = threadIdx.x & 31;
    const int half = lane >> 4;
    const int l16  = lane & 15;
    const int m0 = blockIdx.x * 64;
    const int n0 = (blockIdx.y * 4 + wave) * 64;
    if (m0 >= M) return;

    v8f acc[4][4] = {};
    // single per-lane base pointer; row-block offsets are compile-time immediates
    const unsigned short* Abase = A + (size_t)(m0 + l16) * DDIM + half * 16;
    const unsigned short* Wbase = W + (size_t)(n0 + l16) * DDIM + half * 16;

    for (int k = 0; k < DDIM; k += 32) {
        v16bf a[4], b[4];
#pragma unroll
        for (int i = 0; i < 4; ++i) a[i] = *(const v16bf*)(Abase + i * (16 * DDIM) + k);
#pragma unroll
        for (int j = 0; j < 4; ++j) b[j] = *(const v16bf*)(Wbase + j * (16 * DDIM) + k);
#pragma unroll
        for (int i = 0; i < 4; ++i)
#pragma unroll
            for (int j = 0; j < 4; ++j)
                acc[i][j] = __builtin_amdgcn_wmma_f32_16x16x32_bf16(
                    false, a[i], false, b[j], (short)0, acc[i][j], false, false);
    }

#pragma unroll
    for (int i = 0; i < 4; ++i) {
        int rowb = m0 + 16 * i + half * 8;
#pragma unroll
        for (int j = 0; j < 4; ++j) {
            int col = n0 + 16 * j + l16;
            float bv = bias[col];
#pragma unroll
            for (int v = 0; v < 8; ++v) {
                float val = acc[i][j][v] + bv;
                size_t off = (size_t)(rowb + v) * DDIM + col;
                if (out_bf) out_bf[off] = f32_to_bf16(val);
                else        out_f32[off] = val;
            }
        }
    }
}

// ---------------------------------------------------------------- interleaved RoPE on bf16 [B,S,H,HD]
__global__ void rope_bf16(unsigned short* __restrict__ x, int total_pairs) {
    int i = blockIdx.x * 256 + threadIdx.x;
    if (i >= total_pairs) return;
    int p = i & 31;                     // pair idx within head (HD/2 = 32)
    int s = (i >> 9) & (SQ - 1);        // [b,s,h,p] : h=4 bits, p=5 bits
    float freq = __expf(-(float)p * 0.28782313662425572f);  // ln(10000)/32
    float ang  = (float)s * freq;
    float c = __cosf(ang), sn = __sinf(ang);
    unsigned short* ptr = x + (size_t)i * 2;
    float xe = bf16_to_f32(ptr[0]);
    float xo = bf16_to_f32(ptr[1]);
    ptr[0] = f32_to_bf16(xe * c - xo * sn);
    ptr[1] = f32_to_bf16(xe * sn + xo * c);
}

// ---------------------------------------------------------------- V [B,SKV,H,HD] -> Vt [B,H,HD,SKV]
__global__ void transpose_v(const unsigned short* __restrict__ V, unsigned short* __restrict__ Vt) {
    int i = blockIdx.x * 256 + threadIdx.x;       // over B*H*HD*SKV = 4M
    int s = i & (SKV - 1);
    int d = (i >> 11) & (HD - 1);
    int h = (i >> 17) & (NH - 1);
    int b = i >> 21;
    Vt[i] = V[((size_t)(b * SKV + s) * NH + h) * HD + d];
}

// ---------------------------------------------------------------- flash attention: 1 wave = 16 q rows, kv tiles of 32
__global__ void __launch_bounds__(128, 1)
attn_kernel(const unsigned short* __restrict__ Qb,   // [B,SQ,1024] bf16 (roped)
            const unsigned short* __restrict__ Kb,   // [B,SKV,1024] bf16 (roped)
            const unsigned short* __restrict__ Vt,   // [B,H,HD,SKV] bf16
            unsigned short* __restrict__ ctx) {      // [B,SQ,1024] bf16
    __shared__ unsigned short pLds[4][16 * 32];
    const int wave = threadIdx.x >> 5;
    const int lane = threadIdx.x & 31;
    const int half = lane >> 4;
    const int l16  = lane & 15;

    int strip = blockIdx.x * 4 + wave;            // B*H*(SQ/16) strips
    int q0 = (strip & (SQ / 16 - 1)) * 16;
    int h  = (strip >> 7) & (NH - 1);
    int b  = strip >> 11;

    // Q fragments for 16 rows x 64 hd (two 16x32 A-frags), held in registers
    const unsigned short* Qbase = Qb + (size_t)(b * SQ + q0 + l16) * DDIM + h * HD + half * 16;
    v16bf qf0 = *(const v16bf*)(Qbase);
    v16bf qf1 = *(const v16bf*)(Qbase + 32);

    const unsigned short* Kbase = Kb + (size_t)b * SKV * DDIM + h * HD + half * 16
                                + (size_t)l16 * DDIM;
    const unsigned short* Vbase = Vt + (size_t)(b * NH + h) * HD * SKV + (size_t)half * 16
                                + (size_t)l16 * SKV;

    float mi[8], li[8];
#pragma unroll
    for (int v = 0; v < 8; ++v) { mi[v] = -1e30f; li[v] = 0.f; }
    v8f O[4] = {};
    unsigned short* pW = pLds[wave];

    for (int t = 0; t < SKV; t += 32) {
        const unsigned short* kr = Kbase + (size_t)t * DDIM;
        // kv rows t+l16 and t+16+l16; hd halves at +0 / +32 (immediates)
        v16bf k0a = *(const v16bf*)(kr);
        v16bf k0b = *(const v16bf*)(kr + 32);
        v16bf k1a = *(const v16bf*)(kr + 16 * DDIM);
        v16bf k1b = *(const v16bf*)(kr + 16 * DDIM + 32);

        v8f s0 = {}, s1 = {};
        s0 = __builtin_amdgcn_wmma_f32_16x16x32_bf16(false, qf0, false, k0a, (short)0, s0, false, false);
        s0 = __builtin_amdgcn_wmma_f32_16x16x32_bf16(false, qf1, false, k0b, (short)0, s0, false, false);
        s1 = __builtin_amdgcn_wmma_f32_16x16x32_bf16(false, qf0, false, k1a, (short)0, s1, false, false);
        s1 = __builtin_amdgcn_wmma_f32_16x16x32_bf16(false, qf1, false, k1b, (short)0, s1, false, false);

#pragma unroll
        for (int v = 0; v < 8; ++v) {
            s0[v] *= 0.125f;                       // 1/sqrt(64)
            s1[v] *= 0.125f;
            // row max across the 16 lanes holding this row
            float rm = fmaxf(s0[v], s1[v]);
            rm = fmaxf(rm, __shfl_xor(rm, 1, 16));
            rm = fmaxf(rm, __shfl_xor(rm, 2, 16));
            rm = fmaxf(rm, __shfl_xor(rm, 4, 16));
            rm = fmaxf(rm, __shfl_xor(rm, 8, 16));
            float mnew  = fmaxf(mi[v], rm);
            float alpha = __expf(mi[v] - mnew);
            mi[v] = mnew;
            float p0 = __expf(s0[v] - mnew);
            float p1 = __expf(s1[v] - mnew);
            float rs = p0 + p1;
            rs += __shfl_xor(rs, 1, 16);
            rs += __shfl_xor(rs, 2, 16);
            rs += __shfl_xor(rs, 4, 16);
            rs += __shfl_xor(rs, 8, 16);
            li[v] = li[v] * alpha + rs;
#pragma unroll
            for (int n = 0; n < 4; ++n) O[n][v] *= alpha;
            // C-layout -> LDS (row-major 16x32 bf16 P tile, per-wave region)
            int row = v + 8 * half;
            pW[row * 32 + l16]      = f32_to_bf16(p0);
            pW[row * 32 + l16 + 16] = f32_to_bf16(p1);
        }
        asm volatile("s_wait_dscnt 0" ::: "memory");
        // reload P in A-fragment layout: lane row = l16, k-half from lane group
        v16bf pf = *(const v16bf*)(pW + l16 * 32 + half * 16);
#pragma unroll
        for (int n = 0; n < 4; ++n) {
            v16bf vf = *(const v16bf*)(Vbase + (size_t)(n * 16) * SKV + t);
            O[n] = __builtin_amdgcn_wmma_f32_16x16x32_bf16(false, pf, false, vf, (short)0, O[n], false, false);
        }
    }

#pragma unroll
    for (int n = 0; n < 4; ++n)
#pragma unroll
        for (int v = 0; v < 8; ++v) {
            float val = O[n][v] / li[v];
            int row = q0 + v + 8 * half;
            int col = h * HD + n * 16 + l16;
            ctx[(size_t)(b * SQ + row) * DDIM + col] = f32_to_bf16(val);
        }
}

// ----------------------------------------------------------------
extern "C" void kernel_launch(void* const* d_in, const int* in_sizes, int n_in,
                              void* d_out, int out_size, void* d_ws, size_t ws_size,
                              hipStream_t stream) {
    const float* x_q  = (const float*)d_in[0];
    const float* x_kv = (const float*)d_in[1];
    const float* wq   = (const float*)d_in[2];
    const float* bq   = (const float*)d_in[3];
    const float* wk   = (const float*)d_in[4];
    const float* bk   = (const float*)d_in[5];
    const float* wv   = (const float*)d_in[6];
    const float* bv   = (const float*)d_in[7];
    const float* wo   = (const float*)d_in[8];
    const float* bo   = (const float*)d_in[9];
    float* out = (float*)d_out;
    (void)in_sizes; (void)n_in; (void)out_size; (void)ws_size;

    char* ws = (char*)d_ws;
    size_t off = 0;
    auto carve = [&](size_t nelem) -> unsigned short* {
        unsigned short* p = (unsigned short*)(ws + off);
        off += ((nelem * 2) + 255) & ~(size_t)255;
        return p;
    };
    const int M = BATCH * SQ;                       // 4096
    unsigned short* xq_bf  = carve((size_t)M * DDIM);
    unsigned short* xkv_bf = carve((size_t)M * DDIM);
    unsigned short* wq_bf  = carve((size_t)DDIM * DDIM);
    unsigned short* wk_bf  = carve((size_t)DDIM * DDIM);
    unsigned short* wv_bf  = carve((size_t)DDIM * DDIM);
    unsigned short* wo_bf  = carve((size_t)DDIM * DDIM);
    unsigned short* Qbuf   = carve((size_t)M * DDIM);
    unsigned short* Kbuf   = carve((size_t)M * DDIM);
    unsigned short* Vbuf   = carve((size_t)M * DDIM);
    unsigned short* Vt     = carve((size_t)M * DDIM);
    unsigned short* ctxb   = carve((size_t)M * DDIM);

    const int nAct = M * DDIM;                      // 4,194,304
    const int nW   = DDIM * DDIM;                   // 1,048,576
    cvt_f32_bf16<<<(nAct + 255) / 256, 256, 0, stream>>>(x_q,  xq_bf,  nAct);
    cvt_f32_bf16<<<(nAct + 255) / 256, 256, 0, stream>>>(x_kv, xkv_bf, nAct);
    cvt_f32_bf16<<<(nW + 255) / 256, 256, 0, stream>>>(wq, wq_bf, nW);
    cvt_f32_bf16<<<(nW + 255) / 256, 256, 0, stream>>>(wk, wk_bf, nW);
    cvt_f32_bf16<<<(nW + 255) / 256, 256, 0, stream>>>(wv, wv_bf, nW);
    cvt_f32_bf16<<<(nW + 255) / 256, 256, 0, stream>>>(wo, wo_bf, nW);

    dim3 gg(M / 64, DDIM / 256);                    // (64, 4), 4 waves/block
    gemm_bf16<<<gg, 128, 0, stream>>>(xq_bf,  wq_bf, bq, Qbuf, nullptr, M);
    gemm_bf16<<<gg, 128, 0, stream>>>(xkv_bf, wk_bf, bk, Kbuf, nullptr, M);
    gemm_bf16<<<gg, 128, 0, stream>>>(xkv_bf, wv_bf, bv, Vbuf, nullptr, M);

    const int pairs = M * NH * (HD / 2);            // 2,097,152
    rope_bf16<<<(pairs + 255) / 256, 256, 0, stream>>>(Qbuf, pairs);
    rope_bf16<<<(pairs + 255) / 256, 256, 0, stream>>>(Kbuf, pairs);

    transpose_v<<<nAct / 256, 256, 0, stream>>>(Vbuf, Vt);

    const int strips = BATCH * NH * (SQ / 16);      // 4096
    attn_kernel<<<strips / 4, 128, 0, stream>>>(Qbuf, Kbuf, Vt, ctxb);

    gemm_bf16<<<gg, 128, 0, stream>>>(ctxb, wo_bf, bo, nullptr, out, M);
}
